// GraphSAGELinkPredictor_33517924778074
// MI455X (gfx1250) — matrix operand order repaired and checked by reading, not compiled, not run on hardware
//
#include <hip/hip_runtime.h>
#include <hip/hip_fp16.h>

typedef __attribute__((ext_vector_type(16))) _Float16 v16h;
typedef __attribute__((ext_vector_type(8)))  float    v8f;

#define N_NODES 50000
#define N_EDGES 800000
#define IN_CH   64
#define HID     128

// ---------------------------------------------------------------- utilities

__global__ void zero_f32_kernel(float* __restrict__ p, int n) {
    int i = blockIdx.x * blockDim.x + threadIdx.x;
    if (i < n) p[i] = 0.0f;
}

__global__ void degree_kernel(const int* __restrict__ dst, float* __restrict__ deg, int E) {
    int e = blockIdx.x * blockDim.x + threadIdx.x;
    if (e < E) atomicAdd(&deg[dst[e]], 1.0f);
}

// Split fp32 weights into hi/lo f16 pairs: w ~= hi + lo  (|lo| <= ulp(hi)/2)
__global__ void split_weights_kernel(const float* __restrict__ W,
                                     _Float16* __restrict__ hi,
                                     _Float16* __restrict__ lo, int n) {
    int i = blockIdx.x * blockDim.x + threadIdx.x;
    if (i < n) {
        float w = W[i];
        _Float16 h = (_Float16)w;
        hi[i] = h;
        lo[i] = (_Float16)(w - (float)h);
    }
}

// ------------------------------------------------------- edge scatter (sum)
// One thread handles 4 channels of one edge. CH_SHIFT = log2(K/4).
template <int CH_SHIFT>
__global__ void scatter_add_kernel(const float* __restrict__ hin,
                                   const int* __restrict__ src,
                                   const int* __restrict__ dst,
                                   float* __restrict__ agg, int E) {
    const int K = 4 << CH_SHIFT;
    int t = blockIdx.x * blockDim.x + threadIdx.x;
    int e = t >> CH_SHIFT;
    if (e >= E) return;
    int c = (t & ((1 << CH_SHIFT) - 1)) << 2;
    int s = src[e];
    int d = dst[e];
    const float4 v = *reinterpret_cast<const float4*>(hin + (size_t)s * K + c);
    float* ap = agg + (size_t)d * K + c;
    atomicAdd(ap + 0, v.x);
    atomicAdd(ap + 1, v.y);
    atomicAdd(ap + 2, v.z);
    atomicAdd(ap + 3, v.w);
}

// ------------------------------------------- fused SAGE layer (WMMA GEMM)
// out[n,:] = L2norm( (agg[n,:]/max(deg,1)) @ Wl^T + bl + hin[n,:] @ Wr^T ), opt ReLU
// One wave (32 lanes) computes a full 16-row x 128-col output block.
// A (16x32 f16): lane m=lane&15 is row M; halves j -> K=kb+j (16 contiguous),
//                kb = 16*(lane>>4)  (matches B striping)
// B (32x16 f16): lane holds column N=lane&15; 16 contiguous K halves starting at
//                16*(lane>>4)     (same k order as A)
// C/D (16x16 f32): VGPR r, lane -> M = r + 8*(lane>>4), N = lane&15
template <int K, bool RELU>
__global__ __launch_bounds__(256) void sage_layer_kernel(
    const float* hin, const float* __restrict__ agg, const float* __restrict__ deg,
    const _Float16* __restrict__ WlHi, const _Float16* __restrict__ WlLo,
    const _Float16* __restrict__ WrHi, const _Float16* __restrict__ WrLo,
    const float* __restrict__ bias, float* out, int n) {
    const int lane = threadIdx.x & 31;
    const int wave = threadIdx.x >> 5;
    const int tile = blockIdx.x * 8 + wave;
    const int row_base = tile << 4;
    if (row_base >= n) return;  // wave-uniform: EXEC stays all-ones for WMMA

    const int m = lane & 15;
    const int hsel = lane >> 4;
    const int arow = row_base + m;

    const float invdeg = 1.0f / fmaxf(deg[arow], 1.0f);
    const float* aggRow = agg + (size_t)arow * K;
    const float* hRow = hin + (size_t)arow * K;

    v8f acc[8];
#pragma unroll
    for (int t = 0; t < 8; ++t)
#pragma unroll
        for (int r = 0; r < 8; ++r) acc[t][r] = 0.0f;

#pragma unroll
    for (int ks = 0; ks < K / 32; ++ks) {
        const int kb = ks * 32 + hsel * 16;
        v16h aMh, aMl, aHh, aHl;
#pragma unroll
        for (int j = 0; j < 8; ++j) {
            float vm0 = aggRow[kb + j] * invdeg;
            float vm1 = aggRow[kb + 8 + j] * invdeg;
            _Float16 p0 = (_Float16)vm0;
            _Float16 p1 = (_Float16)vm1;
            aMh[j] = p0;     aMl[j] = (_Float16)(vm0 - (float)p0);
            aMh[8 + j] = p1; aMl[8 + j] = (_Float16)(vm1 - (float)p1);
            float vh0 = hRow[kb + j];
            float vh1 = hRow[kb + 8 + j];
            _Float16 q0 = (_Float16)vh0;
            _Float16 q1 = (_Float16)vh1;
            aHh[j] = q0;     aHl[j] = (_Float16)(vh0 - (float)q0);
            aHh[8 + j] = q1; aHl[8 + j] = (_Float16)(vh1 - (float)q1);
        }
        const int bk = ks * 32 + hsel * 16;
#pragma unroll
        for (int t = 0; t < 8; ++t) {
            const size_t woff = (size_t)(t * 16 + m) * K + bk;
            v16h blh = *(const v16h*)(WlHi + woff);
            v16h bll = *(const v16h*)(WlLo + woff);
            v16h brh = *(const v16h*)(WrHi + woff);
            v16h brl = *(const v16h*)(WrLo + woff);
            // mean @ Wl^T  (hi*hi + lo*hi + hi*lo)
            acc[t] = __builtin_amdgcn_wmma_f32_16x16x32_f16(false, aMh, false, blh, (short)0, acc[t], false, false);
            acc[t] = __builtin_amdgcn_wmma_f32_16x16x32_f16(false, aMl, false, blh, (short)0, acc[t], false, false);
            acc[t] = __builtin_amdgcn_wmma_f32_16x16x32_f16(false, aMh, false, bll, (short)0, acc[t], false, false);
            // h @ Wr^T
            acc[t] = __builtin_amdgcn_wmma_f32_16x16x32_f16(false, aHh, false, brh, (short)0, acc[t], false, false);
            acc[t] = __builtin_amdgcn_wmma_f32_16x16x32_f16(false, aHl, false, brh, (short)0, acc[t], false, false);
            acc[t] = __builtin_amdgcn_wmma_f32_16x16x32_f16(false, aHh, false, brl, (short)0, acc[t], false, false);
        }
    }

    // bias (per output column N = t*16 + m)
#pragma unroll
    for (int t = 0; t < 8; ++t) {
        float bv = bias[t * 16 + m];
#pragma unroll
        for (int r = 0; r < 8; ++r) acc[t][r] += bv;
    }

    // Fused row L2-normalize (+ReLU). Row M = hsel*8 + r lives in VGPR r across
    // the 16 lanes of this half -> xor-shuffle reduce within the half.
#pragma unroll
    for (int r = 0; r < 8; ++r) {
        float ss = 0.0f;
#pragma unroll
        for (int t = 0; t < 8; ++t) {
            float v = acc[t][r];
            ss += v * v;
        }
        ss += __shfl_xor(ss, 1, 32);
        ss += __shfl_xor(ss, 2, 32);
        ss += __shfl_xor(ss, 4, 32);
        ss += __shfl_xor(ss, 8, 32);
        float scale = 1.0f / fmaxf(sqrtf(ss), 1e-12f);
#pragma unroll
        for (int t = 0; t < 8; ++t) {
            float v = acc[t][r] * scale;
            if (RELU) v = fmaxf(v, 0.0f);
            acc[t][r] = v;
        }
    }

    // store (out row M = row_base + hsel*8 + r, col N = t*16 + m)
#pragma unroll
    for (int r = 0; r < 8; ++r) {
        float* op = out + (size_t)(row_base + hsel * 8 + r) * HID;
#pragma unroll
        for (int t = 0; t < 8; ++t) op[t * 16 + m] = acc[t][r];
    }
}

// ---------------------------------------------------------------- launcher

static inline int cdiv(int a, int b) { return (a + b - 1) / b; }

extern "C" void kernel_launch(void* const* d_in, const int* in_sizes, int n_in,
                              void* d_out, int out_size, void* d_ws, size_t ws_size,
                              hipStream_t stream) {
    const float* x = (const float*)d_in[0];
    const int* ei = (const int*)d_in[1];
    const int* src = ei;
    const int* dst = ei + N_EDGES;
    const float* Wl0 = (const float*)d_in[2];
    const float* bl0 = (const float*)d_in[3];
    const float* Wr0 = (const float*)d_in[4];
    const float* Wl1 = (const float*)d_in[5];
    const float* bl1 = (const float*)d_in[6];
    const float* Wr1 = (const float*)d_in[7];
    const float* Wl2 = (const float*)d_in[8];
    const float* bl2 = (const float*)d_in[9];
    const float* Wr2 = (const float*)d_in[10];
    float* out = (float*)d_out;

    // workspace carve (256B aligned slices)
    char* ws = (char*)d_ws;
    size_t off = 0;
    auto carve = [&](size_t bytes) -> char* {
        char* p = ws + off;
        off += (bytes + 255) & ~(size_t)255;
        return p;
    };
    float* agg = (float*)carve((size_t)N_NODES * HID * 4);
    float* h = (float*)carve((size_t)N_NODES * HID * 4);
    float* deg = (float*)carve((size_t)N_NODES * 4);
    _Float16* Wl0h = (_Float16*)carve((size_t)HID * IN_CH * 2);
    _Float16* Wl0l = (_Float16*)carve((size_t)HID * IN_CH * 2);
    _Float16* Wr0h = (_Float16*)carve((size_t)HID * IN_CH * 2);
    _Float16* Wr0l = (_Float16*)carve((size_t)HID * IN_CH * 2);
    _Float16* Wl1h = (_Float16*)carve((size_t)HID * HID * 2);
    _Float16* Wl1l = (_Float16*)carve((size_t)HID * HID * 2);
    _Float16* Wr1h = (_Float16*)carve((size_t)HID * HID * 2);
    _Float16* Wr1l = (_Float16*)carve((size_t)HID * HID * 2);
    _Float16* Wl2h = (_Float16*)carve((size_t)HID * HID * 2);
    _Float16* Wl2l = (_Float16*)carve((size_t)HID * HID * 2);
    _Float16* Wr2h = (_Float16*)carve((size_t)HID * HID * 2);
    _Float16* Wr2l = (_Float16*)carve((size_t)HID * HID * 2);

    const int TPB = 256;

    // degree (shared by all 3 layers)
    zero_f32_kernel<<<cdiv(N_NODES, TPB), TPB, 0, stream>>>(deg, N_NODES);
    degree_kernel<<<cdiv(N_EDGES, TPB), TPB, 0, stream>>>(dst, deg, N_EDGES);

    // weight splits
    split_weights_kernel<<<cdiv(HID * IN_CH, TPB), TPB, 0, stream>>>(Wl0, Wl0h, Wl0l, HID * IN_CH);
    split_weights_kernel<<<cdiv(HID * IN_CH, TPB), TPB, 0, stream>>>(Wr0, Wr0h, Wr0l, HID * IN_CH);
    split_weights_kernel<<<cdiv(HID * HID, TPB), TPB, 0, stream>>>(Wl1, Wl1h, Wl1l, HID * HID);
    split_weights_kernel<<<cdiv(HID * HID, TPB), TPB, 0, stream>>>(Wr1, Wr1h, Wr1l, HID * HID);
    split_weights_kernel<<<cdiv(HID * HID, TPB), TPB, 0, stream>>>(Wl2, Wl2h, Wl2l, HID * HID);
    split_weights_kernel<<<cdiv(HID * HID, TPB), TPB, 0, stream>>>(Wr2, Wr2h, Wr2l, HID * HID);

    const int gemm_blocks = cdiv(N_NODES / 16, 8);

    // ---- layer 0: x (N x 64) -> h (N x 128), ReLU
    zero_f32_kernel<<<cdiv(N_NODES * IN_CH, TPB), TPB, 0, stream>>>(agg, N_NODES * IN_CH);
    scatter_add_kernel<4><<<cdiv(N_EDGES * (IN_CH / 4), TPB), TPB, 0, stream>>>(x, src, dst, agg, N_EDGES);
    sage_layer_kernel<IN_CH, true><<<gemm_blocks, TPB, 0, stream>>>(
        x, agg, deg, Wl0h, Wl0l, Wr0h, Wr0l, bl0, h, N_NODES);

    // ---- layer 1: h -> h (in place), ReLU
    zero_f32_kernel<<<cdiv(N_NODES * HID, TPB), TPB, 0, stream>>>(agg, N_NODES * HID);
    scatter_add_kernel<5><<<cdiv(N_EDGES * (HID / 4), TPB), TPB, 0, stream>>>(h, src, dst, agg, N_EDGES);
    sage_layer_kernel<HID, true><<<gemm_blocks, TPB, 0, stream>>>(
        h, agg, deg, Wl1h, Wl1l, Wr1h, Wr1l, bl1, h, N_NODES);

    // ---- layer 2: h -> out, no ReLU
    zero_f32_kernel<<<cdiv(N_NODES * HID, TPB), TPB, 0, stream>>>(agg, N_NODES * HID);
    scatter_add_kernel<5><<<cdiv(N_EDGES * (HID / 4), TPB), TPB, 0, stream>>>(h, src, dst, agg, N_EDGES);
    sage_layer_kernel<HID, false><<<gemm_blocks, TPB, 0, stream>>>(
        h, agg, deg, Wl2h, Wl2l, Wr2h, Wr2l, bl2, out, N_NODES);
}
